// GAT_54992761258610
// MI455X (gfx1250) — matrix operand order repaired and checked by reading, not compiled, not run on hardware
//
#include <hip/hip_runtime.h>
#include <hip/hip_bf16.h>

typedef __attribute__((ext_vector_type(16))) _Float16 v16h;
typedef __attribute__((ext_vector_type(8)))  float    v8f;

#define HEADS 8
#define CH 16
#define HC 128
#define NEG_SLOPE 0.2f

// ---- float <-> monotonic-unsigned ordinal (for atomicMax on f32) ----
__device__ __forceinline__ unsigned f_ord(float f) {
    unsigned u = __float_as_uint(f);
    return (u >> 31) ? ~u : (u | 0x80000000u);
}
__device__ __forceinline__ float ord_f(unsigned u) {
    return (u >> 31) ? __uint_as_float(u & 0x7fffffffu) : __uint_as_float(~u);
}
__device__ __forceinline__ float leaky(float v) {
    return v > 0.0f ? v : v * NEG_SLOPE;
}

// ============================================================
// Kernel 1: h = x @ W via v_wmma_f32_16x16x32_f16
// block = 256 threads (8 waves); block b covers nodes [16b,16b+16)
// wave w computes output columns [16w,16w+16) == head w.
// ============================================================
__global__ void gat55_h_wmma(const float* __restrict__ x,
                             const float* __restrict__ W,
                             float* __restrict__ h, int N) {
    const int wave = threadIdx.x >> 5;   // 0..7 -> column tile / head
    const int lane = threadIdx.x & 31;
    const int g    = lane >> 4;          // K-group 0/1
    const int r    = lane & 15;          // A: row M ; B/D: col N
    const int node0 = blockIdx.x * 16;
    const int col0  = wave * 16;

    // A fragment (16x32 f16, K=16 real + 16 zero pad):
    // lane (g, M=r): g==0: elem e in 0..15 -> x[node0+r, e]; g==1 -> 0
    int arow = node0 + r;
    if (arow >= N) arow = N - 1;  // clamp (stores are guarded)
    const float* xrow = x + (size_t)arow * 16;
    v16h a = {};
    if (g == 0) {
#pragma unroll
    for (int e = 0; e < 16; ++e) a[e] = (_Float16)xrow[e];
    }

    // B fragment (32x16 f16): lane (g, N=r): elem i -> W[g*16+i, col0+r]
    // g==1 rows are the K=16..31 zero pad.
    v16h b = {};
    if (g == 0) {
#pragma unroll
        for (int i = 0; i < 16; ++i) b[i] = (_Float16)W[i * HC + col0 + r];
    }

    v8f c = {};
    c = __builtin_amdgcn_wmma_f32_16x16x32_f16(
        /*neg_a=*/false, a, /*neg_b=*/false, b,
        /*c_mod=*/(short)0, c, /*reuse_a=*/false, /*reuse_b=*/false);

    // D: lane (g, N=r): elem m -> h[node0 + g*8 + m, col0 + r]
#pragma unroll
    for (int m = 0; m < 8; ++m) {
        int row = node0 + g * 8 + m;
        if (row < N) h[(size_t)row * HC + col0 + r] = c[m];
    }
}

// ============================================================
// Kernel 2: a_src[n,hd] = <h[n,hd,:], att_src[hd,:]>, same for dst
// ============================================================
__global__ void gat55_attn(const float* __restrict__ h,
                           const float* __restrict__ att_src,
                           const float* __restrict__ att_dst,
                           float* __restrict__ a_src,
                           float* __restrict__ a_dst, int NH) {
    int i = blockIdx.x * blockDim.x + threadIdx.x;
    if (i >= NH) return;
    int n = i >> 3, hd = i & 7;
    const float* hp = h + (size_t)n * HC + hd * CH;
    const float* as = att_src + hd * CH;
    const float* ad = att_dst + hd * CH;
    float s = 0.f, d = 0.f;
#pragma unroll
    for (int c = 0; c < CH; ++c) {
        float v = hp[c];
        s += v * as[c];
        d += v * ad[c];
    }
    a_src[i] = s;
    a_dst[i] = d;
}

// ============================================================
// Kernel 3: segment max over edges by dst (ordinal-encoded atomicMax)
// one thread per (edge, head); edges >= E are self loops.
// ============================================================
__global__ void gat55_edge_max(const int* __restrict__ srcs,
                               const int* __restrict__ dsts,
                               const float* __restrict__ a_src,
                               const float* __restrict__ a_dst,
                               unsigned* __restrict__ emax_ord,
                               int E, int N) {
    int i = blockIdx.x * blockDim.x + threadIdx.x;
    int tot = (E + N) * HEADS;
    if (i >= tot) return;
    int e = i >> 3, hd = i & 7;
    int s = (e < E) ? srcs[e] : (e - E);
    int d = (e < E) ? dsts[e] : (e - E);
    float v = leaky(a_src[s * HEADS + hd] + a_dst[d * HEADS + hd]);
    atomicMax(&emax_ord[d * HEADS + hd], f_ord(v));
}

// ============================================================
// Kernel 4: denom[dst,hd] += exp(e - emax[dst,hd])
// ============================================================
__global__ void gat55_edge_denom(const int* __restrict__ srcs,
                                 const int* __restrict__ dsts,
                                 const float* __restrict__ a_src,
                                 const float* __restrict__ a_dst,
                                 const unsigned* __restrict__ emax_ord,
                                 float* __restrict__ denom,
                                 int E, int N) {
    int i = blockIdx.x * blockDim.x + threadIdx.x;
    int tot = (E + N) * HEADS;
    if (i >= tot) return;
    int e = i >> 3, hd = i & 7;
    int s = (e < E) ? srcs[e] : (e - E);
    int d = (e < E) ? dsts[e] : (e - E);
    float v = leaky(a_src[s * HEADS + hd] + a_dst[d * HEADS + hd]);
    float m = ord_f(emax_ord[d * HEADS + hd]);
    atomicAdd(&denom[d * HEADS + hd], __expf(v - m));
}

// ============================================================
// Kernel 5: acc[dst, :] += alpha * h[src, :]
// 256 threads / block = 2 edges; thread t in 0..127 handles channel t
// (head = t/16). Gathers & atomics fully coalesced; everything L2-resident.
// ============================================================
__global__ void gat55_scatter(const int* __restrict__ srcs,
                              const int* __restrict__ dsts,
                              const float* __restrict__ a_src,
                              const float* __restrict__ a_dst,
                              const unsigned* __restrict__ emax_ord,
                              const float* __restrict__ denom,
                              const float* __restrict__ h,
                              float* __restrict__ acc,
                              int E, int N) {
    int e = blockIdx.x * 2 + (threadIdx.x >> 7);
    if (e >= E + N) return;
    int t  = threadIdx.x & 127;
    int hd = t >> 4;
    int s = (e < E) ? srcs[e] : (e - E);
    int d = (e < E) ? dsts[e] : (e - E);
    float v = leaky(a_src[s * HEADS + hd] + a_dst[d * HEADS + hd]);
    float m = ord_f(emax_ord[d * HEADS + hd]);
    float alpha = __expf(v - m) / (denom[d * HEADS + hd] + 1e-16f);
    atomicAdd(&acc[(size_t)d * HC + t], alpha * h[(size_t)s * HC + t]);
}

// ============================================================
// Kernel 6: out = log_softmax(relu(acc + bias) @ fc_w + fc_b)
// one wave32 per node; lane owns 4 channels; shfl_xor reduction.
// ============================================================
__global__ void gat55_head(const float* __restrict__ acc,
                           const float* __restrict__ bias,
                           const float* __restrict__ fc_w,
                           const float* __restrict__ fc_b,
                           float* __restrict__ out, int N) {
    int gid  = blockIdx.x * blockDim.x + threadIdx.x;
    int node = gid >> 5;
    int lane = gid & 31;
    if (node >= N) return;

    float v[4];
#pragma unroll
    for (int j = 0; j < 4; ++j) {
        int ch = lane * 4 + j;
        float t = acc[(size_t)node * HC + ch] + bias[ch];
        v[j] = t > 0.f ? t : 0.f;
    }
    float logit[5];
#pragma unroll
    for (int o = 0; o < 5; ++o) {
        float ssum = 0.f;
#pragma unroll
        for (int j = 0; j < 4; ++j) ssum += v[j] * fc_w[(lane * 4 + j) * 5 + o];
        for (int off = 16; off > 0; off >>= 1) ssum += __shfl_xor(ssum, off, 32);
        logit[o] = ssum + fc_b[o];
    }
    if (lane == 0) {
        float mx = logit[0];
#pragma unroll
        for (int o = 1; o < 5; ++o) mx = fmaxf(mx, logit[o]);
        float sum = 0.f;
#pragma unroll
        for (int o = 0; o < 5; ++o) sum += __expf(logit[o] - mx);
        float lse = mx + __logf(sum);
#pragma unroll
        for (int o = 0; o < 5; ++o) out[(size_t)node * 5 + o] = logit[o] - lse;
    }
}

extern "C" void kernel_launch(void* const* d_in, const int* in_sizes, int n_in,
                              void* d_out, int out_size, void* d_ws, size_t ws_size,
                              hipStream_t stream) {
    const float* x       = (const float*)d_in[0];
    const int*   ei      = (const int*)  d_in[1];
    const float* W       = (const float*)d_in[2];
    const float* att_src = (const float*)d_in[3];
    const float* att_dst = (const float*)d_in[4];
    const float* bias    = (const float*)d_in[5];
    const float* fc_w    = (const float*)d_in[6];
    const float* fc_b    = (const float*)d_in[7];
    float* out = (float*)d_out;

    const int N = in_sizes[0] / 16;
    const int E = in_sizes[1] / 2;
    const int* srcs = ei;
    const int* dsts = ei + E;

    // workspace layout (floats)
    float*    h     = (float*)d_ws;                       // N*128
    float*    a_src = h + (size_t)N * HC;                 // N*8
    float*    a_dst = a_src + (size_t)N * HEADS;          // N*8
    unsigned* emax  = (unsigned*)(a_dst + (size_t)N * HEADS); // N*8
    float*    denom = (float*)(emax + (size_t)N * HEADS); // N*8
    float*    acc   = denom + (size_t)N * HEADS;          // N*128

    hipMemsetAsync(emax,  0, (size_t)N * HEADS * sizeof(unsigned), stream);
    hipMemsetAsync(denom, 0, (size_t)N * HEADS * sizeof(float),    stream);
    hipMemsetAsync(acc,   0, (size_t)N * HC    * sizeof(float),    stream);

    // 1) h = x @ W  (WMMA)
    gat55_h_wmma<<<(N + 15) / 16, 256, 0, stream>>>(x, W, h, N);

    // 2) attention scalars
    int NH = N * HEADS;
    gat55_attn<<<(NH + 255) / 256, 256, 0, stream>>>(h, att_src, att_dst,
                                                     a_src, a_dst, NH);

    // 3/4) segment max then denom over (E + N self loops) edges
    int tot = (E + N) * HEADS;
    gat55_edge_max<<<(tot + 255) / 256, 256, 0, stream>>>(srcs, dsts, a_src, a_dst,
                                                          emax, E, N);
    gat55_edge_denom<<<(tot + 255) / 256, 256, 0, stream>>>(srcs, dsts, a_src, a_dst,
                                                            emax, denom, E, N);

    // 5) weighted scatter (2 edges per 256-thread block)
    int ET = E + N;
    gat55_scatter<<<(ET + 1) / 2, 256, 0, stream>>>(srcs, dsts, a_src, a_dst,
                                                    emax, denom, h, acc, E, N);

    // 6) bias + relu + fc + log_softmax (one wave per node)
    gat55_head<<<((size_t)N * 32 + 255) / 256, 256, 0, stream>>>(acc, bias, fc_w,
                                                                 fc_b, out, N);
}
